// GATHead_7327214207583
// MI455X (gfx1250) — matrix operand order repaired and checked by reading, not compiled, not run on hardware
//
#include <hip/hip_runtime.h>
#include <hip/hip_bf16.h>

#define BB 4
#define NN 4096
#define DD 256

typedef __attribute__((ext_vector_type(16))) __bf16   v16bf;
typedef __attribute__((ext_vector_type(8)))  float    v8f;
typedef __attribute__((ext_vector_type(4)))  int      v4i;
typedef __attribute__((ext_vector_type(8)))  int      v8i;
typedef __attribute__((ext_vector_type(4)))  unsigned v4u;

__device__ __forceinline__ v8f wmma_bf16(v16bf a, v16bf b, v8f c) {
  // D = A(16x32 bf16) * B(32x16 bf16) + C(16x16 f32)
  return __builtin_amdgcn_wmma_f32_16x16x32_bf16(false, a, false, b, (short)0, c, false, false);
}

__device__ __forceinline__ __bf16 bf(float x) { return (__bf16)x; }

// ---- CDNA5 async global->LDS path (ASYNCcnt) ------------------------------
#if defined(__gfx1250__) && __has_builtin(__builtin_amdgcn_global_load_async_to_lds_b128) && \
    __has_builtin(__builtin_amdgcn_s_wait_asynccnt)
#define USE_ASYNC_LDS 1
#else
#define USE_ASYNC_LDS 0
#endif

// ---- CDNA5 Tensor Data Mover path (TENSORcnt) -----------------------------
#if defined(__gfx1250__) && __has_builtin(__builtin_amdgcn_tensor_load_to_lds) && \
    __has_builtin(__builtin_amdgcn_s_wait_tensorcnt)
#define USE_TDM 1
#else
#define USE_TDM 0
#endif

__device__ __forceinline__ void async_g2l_b128(const void* g, void* l) {
#if USE_ASYNC_LDS
  __builtin_amdgcn_global_load_async_to_lds_b128(
      (__attribute__((address_space(1))) v4i*)(g),
      (__attribute__((address_space(3))) v4i*)(l), 0, 0);
#endif
}

__device__ __forceinline__ void wait_async0() {
#if USE_ASYNC_LDS
  __builtin_amdgcn_s_wait_asynccnt(0);
#endif
}

__device__ __forceinline__ void wait_tensor0() {
#if USE_TDM
  __builtin_amdgcn_s_wait_tensorcnt(0);
#endif
}

__device__ __forceinline__ unsigned lds_addr_of(const void* p) {
  return (unsigned)(__SIZE_TYPE__)(__attribute__((address_space(3))) const void*)p;
}

#if USE_TDM
// 2D tile load via Tensor DMA descriptor (ISA cdna5 §8.3/8.4).
// dims/strides in data_size units; pad fields per D# group1 encoding.
__device__ __forceinline__
void tdm_load_2d(unsigned lds_addr, const void* gptr, unsigned dsz_code,
                 unsigned tile_d0, unsigned tile_d1,
                 unsigned tensor_d0, unsigned tensor_d1,
                 unsigned long long d0_stride,
                 unsigned pad_interval, unsigned pad_amount) {
  const unsigned long long ga = (unsigned long long)(__SIZE_TYPE__)gptr;
  v4u g0 = {};
  g0[0] = 1u;                                              // count=1, user D#
  g0[1] = lds_addr;                                        // LDS byte address
  g0[2] = (unsigned)(ga & 0xffffffffu);                    // global_addr[31:0]
  g0[3] = (unsigned)((ga >> 32) & 0x01ffffffu) | (2u << 30); // addr[56:32] | type=2
  v8i g1 = {};
  g1[0] = (int)((dsz_code << 16) | (1u << 20) |            // data_size | pad_enable
                (pad_interval << 22) | (pad_amount << 25)); // wg_mask=0 (no cluster)
  g1[1] = (int)((tensor_d0 & 0xffffu) << 16);              // tensor_dim0[15:0]
  g1[2] = (int)(((tensor_d0 >> 16) & 0xffffu) |            // tensor_dim0[31:16]
                ((tensor_d1 & 0xffffu) << 16));            // tensor_dim1[15:0]
  g1[3] = (int)(((tensor_d1 >> 16) & 0xffffu) |            // tensor_dim1[31:16]
                ((tile_d0 & 0xffffu) << 16));              // tile_dim0
  g1[4] = (int)(tile_d1 & 0xffffu);                        // tile_dim1 (tile_dim2=0)
  g1[5] = (int)(unsigned)(d0_stride & 0xffffffffu);        // dim0_stride[31:0]
  g1[6] = (int)(unsigned)((d0_stride >> 32) & 0xffffu);    // dim0_stride[47:32]
  g1[7] = 0;                                               // dim1_stride unused (2D)
  v4i gz4 = {};
  v8i gz8 = {};
  // 6-arg toolchain form: (g0, g1, g2, g3, <extra group>, cpol)
  __builtin_amdgcn_tensor_load_to_lds(g0, g1, gz4, gz4, gz8, 0);
}
#endif

// ---------------------------------------------------------------------------
// Kernel 1: feats = data @ W1^T, stored transposed in bf16: featsT[b][o][j]
// One 16(row) x 16(o) tile per wave, K=256 in 8 steps of 32 (bf16 WMMA).
// ---------------------------------------------------------------------------
__global__ __launch_bounds__(128)
void k_feats(const float* __restrict__ data, const float* __restrict__ W1,
             __bf16* __restrict__ featsT) {
  __shared__ float sC[4][256];                 // per-wave 16x16 C tile
  const int lane = threadIdx.x & 31;
  const int w    = threadIdx.x >> 5;
  const int li   = lane & 15;
  const int h    = lane >> 4;
  const int t    = blockIdx.x * 4 + w;         // wave-tile id
  const int r0   = (t >> 4) * 16;              // row tile (over B*N rows)
  const int o0   = (t & 15) * 16;              // output-channel tile

  const float* arow = data + (size_t)(r0 + li) * DD;   // A: M=li across lanes
  const float* brow = W1   + (size_t)(o0 + li) * DD;   // B: N=li across lanes

  v8f acc = 0.f;
  #pragma unroll
  for (int ki = 0; ki < 8; ++ki) {
    const int kb = ki * 32;
    float af[16];
    {
      float4 a0 = *(const float4*)(arow + kb + h * 8);
      float4 a1 = *(const float4*)(arow + kb + h * 8 + 4);
      float4 a2 = *(const float4*)(arow + kb + 16 + h * 8);
      float4 a3 = *(const float4*)(arow + kb + 16 + h * 8 + 4);
      af[0]=a0.x; af[1]=a0.y; af[2]=a0.z; af[3]=a0.w;
      af[4]=a1.x; af[5]=a1.y; af[6]=a1.z; af[7]=a1.w;
      af[8]=a2.x; af[9]=a2.y; af[10]=a2.z; af[11]=a2.w;
      af[12]=a3.x; af[13]=a3.y; af[14]=a3.z; af[15]=a3.w;
    }
    float bfl[16];
    {
      float4 b0 = *(const float4*)(brow + kb + h * 8);
      float4 b1 = *(const float4*)(brow + kb + h * 8 + 4);
      float4 b2 = *(const float4*)(brow + kb + 16 + h * 8);
      float4 b3 = *(const float4*)(brow + kb + 16 + h * 8 + 4);
      bfl[0]=b0.x; bfl[1]=b0.y; bfl[2]=b0.z; bfl[3]=b0.w;
      bfl[4]=b1.x; bfl[5]=b1.y; bfl[6]=b1.z; bfl[7]=b1.w;
      bfl[8]=b2.x; bfl[9]=b2.y; bfl[10]=b2.z; bfl[11]=b2.w;
      bfl[12]=b3.x; bfl[13]=b3.y; bfl[14]=b3.z; bfl[15]=b3.w;
    }
    v16bf a, b;
    #pragma unroll
    for (int q = 0; q < 16; ++q) { a[q] = bf(af[q]); b[q] = bf(bfl[q]); }
    acc = wmma_bf16(a, b, acc);
  }
  #pragma unroll
  for (int v = 0; v < 8; ++v)
    sC[w][(v + 8 * h) * 16 + li] = acc[v];
  __syncthreads();
  if (h == 0) {  // lanes 0..15: each packs one o-row of 16 j-values
    const int bId = r0 / NN;
    const int j0  = r0 % NN;
    union { __bf16 hh[16]; uint4 u[2]; } pk;
    #pragma unroll
    for (int jj = 0; jj < 16; ++jj)
      pk.hh[jj] = bf(sC[w][jj * 16 + li]);
    __bf16* dst = featsT + ((size_t)(bId * DD + o0 + li) * NN + j0);
    *(uint4*)(dst)     = pk.u[0];
    *(uint4*)(dst + 8) = pk.u[1];
  }
}

// ---------------------------------------------------------------------------
// Kernel 2: f1[b][j] = sum_o featsT[b][o][j] * W2[o] + b2   (deterministic)
// ---------------------------------------------------------------------------
__global__ __launch_bounds__(256)
void k_f1(const __bf16* __restrict__ featsT, const float* __restrict__ W2,
          const float* __restrict__ b2, float* __restrict__ f1) {
  const int r = blockIdx.x * 256 + threadIdx.x;      // [0, B*N)
  const int b = r >> 12;
  const int j = r & (NN - 1);
  const __bf16* base = featsT + (size_t)b * DD * NN + j;
  float s = 0.f;
  #pragma unroll 8
  for (int o = 0; o < DD; ++o)
    s += (float)base[(size_t)o * NN] * W2[o];
  f1[r] = s + b2[0];
}

// ---------------------------------------------------------------------------
// Kernel 3: fused flash-attention style GAT aggregation.
// 4 waves/WG, wave w owns rows iw..iw+15; J-blocks of 32, double-buffered
// LDS staged via the Tensor Data Mover (wave 0 issues two 2D-tile D#'s per
// j-block; TENSORcnt + barrier), f1 slice via async-to-LDS.
// ---------------------------------------------------------------------------
#define JB   32
#define NJB  (NN / JB)          // 128 j-blocks
#define STRF 40                 // bf16 elems per sFeats row (32 + 16B pad)
#define STRB 36                 // floats per sBias row (32 + 16B pad)

__device__ __forceinline__
void stage_jblock(const __bf16* __restrict__ featsB, const float* __restrict__ bias1i,
                  const float* __restrict__ f1b, int j0, int tid,
                  __bf16* __restrict__ sF, float* __restrict__ sB,
                  float* __restrict__ sf) {
#if USE_TDM
  if (tid < 32) {   // wave 0 issues the DMA descriptors (wave-level, EXEC-ignored)
    // featsT tile: 256 rows x 32 bf16, row=64B=8 qwords -> pad_interval=3 (16 DW),
    // pad 4 DWORDs -> pad_amount=3  (matches STRF = 32+8 bf16)
    tdm_load_2d(lds_addr_of(sF), featsB + j0, /*dsz=2B*/1,
                /*tile*/32, DD, /*tensor*/NN, DD, /*stride*/NN, 3, 3);
    // bias1 tile: 64 rows x 32 f32, row=128B=16 qwords -> pad_interval=4 (32 DW),
    // pad 4 DWORDs -> pad_amount=3  (matches STRB = 32+4 f32)
    tdm_load_2d(lds_addr_of(sB), bias1i + j0, /*dsz=4B*/2,
                /*tile*/32, 64, /*tensor*/NN, 64, /*stride*/NN, 4, 3);
  }
  if (tid < 8)
    async_g2l_b128(f1b + j0 + tid * 4, sf + tid * 4);
#elif USE_ASYNC_LDS
  #pragma unroll
  for (int q = 0; q < 8; ++q) {              // featsT: 256 rows x 32 bf16
    const int fidx = q * 128 + tid;
    const int o = fidx >> 2, part = fidx & 3;
    async_g2l_b128(featsB + (size_t)o * NN + j0 + part * 8, sF + o * STRF + part * 8);
  }
  #pragma unroll
  for (int q = 0; q < 4; ++q) {              // bias1: 64 rows x 32 f32
    const int fidx = q * 128 + tid;
    const int row = fidx >> 3, part = fidx & 7;
    async_g2l_b128(bias1i + (size_t)row * NN + j0 + part * 4, sB + row * STRB + part * 4);
  }
  if (tid < 8)
    async_g2l_b128(f1b + j0 + tid * 4, sf + tid * 4);
#else
  #pragma unroll
  for (int q = 0; q < 8; ++q) {
    const int fidx = q * 128 + tid;
    const int o = fidx >> 2, part = fidx & 3;
    uint4 v = *(const uint4*)(featsB + (size_t)o * NN + j0 + part * 8);
    *(uint4*)(sF + o * STRF + part * 8) = v;
  }
  #pragma unroll
  for (int q = 0; q < 4; ++q) {
    const int fidx = q * 128 + tid;
    const int row = fidx >> 3, part = fidx & 7;
    float4 v = *(const float4*)(bias1i + (size_t)row * NN + j0 + part * 4);
    *(float4*)(sB + row * STRB + part * 4) = v;
  }
  if (tid < 8) {
    float4 v = *(const float4*)(f1b + j0 + tid * 4);
    *(float4*)(sf + tid * 4) = v;
  }
#endif
}

__global__ __launch_bounds__(128)
void k_attn(const float* __restrict__ data, const float* __restrict__ bias1,
            const float* __restrict__ bias2, const __bf16* __restrict__ featsT,
            const float* __restrict__ f1, float* __restrict__ out) {
  __shared__ __bf16 sFeats[2][DD * STRF];   // 2 x 20480 B
  __shared__ float  sBias[2][64 * STRB];    // 2 x  9216 B
  __shared__ float  sF1[2][JB];             // 2 x   128 B

  const int tid  = threadIdx.x;
  const int lane = tid & 31;
  const int w    = tid >> 5;
  const int li   = lane & 15;
  const int h    = lane >> 4;
  const int nIB  = NN / 64;
  const int b    = blockIdx.x / nIB;
  const int i0   = (blockIdx.x % nIB) * 64;
  const int iw   = i0 + w * 16;                      // this wave's first row

  const float   f1i    = f1[b * NN + iw + li];       // A-layout: row = li
  const __bf16* featsB = featsT + (size_t)b * DD * NN;
  const float*  bias1i = bias1 + (size_t)i0 * NN;
  const float*  f1b    = f1 + b * NN;
  const int     rloc   = w * 16 + li;                // row within sBias window

  v8f acc[16];
  #pragma unroll
  for (int nt = 0; nt < 16; ++nt) acc[nt] = 0.f;
  float m_run = -__builtin_inff();
  float l_run = 0.f;

  // prologue: stage j-block 0
  stage_jblock(featsB, bias1i, f1b, 0, tid, sFeats[0], sBias[0], sF1[0]);
  wait_async0();
  wait_tensor0();
  __syncthreads();

  for (int jb = 0; jb < NJB; ++jb) {
    const int cur = jb & 1;
    // prefetch next j-block into the other buffer (overlaps compute below)
    if (jb + 1 < NJB)
      stage_jblock(featsB, bias1i, f1b, (jb + 1) * JB, tid,
                   sFeats[cur ^ 1], sBias[cur ^ 1], sF1[cur ^ 1]);

    const __bf16* sF = sFeats[cur];
    const float*  sB = sBias[cur];
    const float*  sf = sF1[cur];

    // scores in A-matrix layout: element q of lane(li,h) is
    // j-local = h*8 + q + (q<8 ? 0 : 8)
    float s[16];
    #pragma unroll
    for (int q = 0; q < 16; ++q) {
      const int jl = h * 8 + q + ((q < 8) ? 0 : 8);
      float x = f1i + sf[jl];
      x = (x > 0.f) ? x : 0.01f * x;                 // leaky_relu(0.01)
      s[q] = x + sB[rloc * STRB + jl];
    }
    // online softmax: row stats live in lanes li and li+16 (kept equal)
    float mloc = s[0];
    #pragma unroll
    for (int q = 1; q < 16; ++q) mloc = fmaxf(mloc, s[q]);
    mloc = fmaxf(mloc, __shfl_xor(mloc, 16, 32));
    const float m_new = fmaxf(m_run, mloc);
    const float scale = __expf(m_run - m_new);
    float rs = 0.f;
    v16bf a;
    #pragma unroll
    for (int q = 0; q < 16; ++q) {
      const float p = __expf(s[q] - m_new);
      rs += p;
      a[q] = bf(p);
    }
    rs += __shfl_xor(rs, 16, 32);
    l_run = l_run * scale + rs;
    m_run = m_new;
    // broadcast per-row rescale into C layout (acc row = v + 8h)
    float sc[8];
    #pragma unroll
    for (int v = 0; v < 8; ++v) sc[v] = __shfl(scale, 8 * h + v, 32);
    #pragma unroll
    for (int nt = 0; nt < 16; ++nt) {
      #pragma unroll
      for (int v = 0; v < 8; ++v) acc[nt][v] *= sc[v];
      // B fragment: lane(li,h) = featsT[o = nt*16+li][h*8 .. h*8+7, 16+h*8 .. 16+h*8+7]
      union { uint4 u[2]; v16bf v16; } bb;
      const __bf16* src = &sF[(nt * 16 + li) * STRF + h * 8];
      bb.u[0] = *(const uint4*)(src);
      bb.u[1] = *(const uint4*)(src + 16);
      acc[nt] = wmma_bf16(a, bb.v16, acc[nt]);
    }

    wait_async0();      // f1 prefetch complete (this wave's ASYNCcnt)
    wait_tensor0();     // TDM tiles for jb+1 complete (wave 0's TENSORcnt)
    __syncthreads();    // publish staged LDS to all waves
  }

  // epilogue: normalize, bias2, residual
  float rinv = 1.0f / l_run;
  float rv[8];
  #pragma unroll
  for (int v = 0; v < 8; ++v) rv[v] = __shfl(rinv, 8 * h + v, 32);
  #pragma unroll
  for (int nt = 0; nt < 16; ++nt) {
    const int o  = nt * 16 + li;
    const float bz = bias2[o];
    #pragma unroll
    for (int v = 0; v < 8; ++v) {
      const size_t idx = (size_t)(b * NN + iw + v + 8 * h) * DD + o;
      out[idx] = acc[nt][v] * rv[v] + bz + data[idx];
    }
  }
}

// ---------------------------------------------------------------------------
extern "C" void kernel_launch(void* const* d_in, const int* in_sizes, int n_in,
                              void* d_out, int out_size, void* d_ws, size_t ws_size,
                              hipStream_t stream) {
  const float* data  = (const float*)d_in[0];
  const float* bias1 = (const float*)d_in[1];
  const float* W1    = (const float*)d_in[2];
  const float* W2    = (const float*)d_in[3];
  const float* b2    = (const float*)d_in[4];
  const float* bias2 = (const float*)d_in[5];
  float* out = (float*)d_out;

  __bf16* featsT = (__bf16*)d_ws;                       // B*256*N bf16 = 8 MiB
  float*  f1     = (float*)((char*)d_ws + (size_t)BB * DD * NN * sizeof(__bf16));

  const int tiles = (BB * NN / 16) * (DD / 16);
  k_feats<<<tiles / 4, 128, 0, stream>>>(data, W1, featsT);
  k_f1<<<BB * NN / 256, 256, 0, stream>>>(featsT, W2, b2, f1);
  k_attn<<<BB * NN / 64, 128, 0, stream>>>(data, bias1, bias2, featsT, f1, out);
}
